// GCN_GMM_53094385713812
// MI455X (gfx1250) — compile-verified, matrix operand order checked by reading, not run on hardware
//
#include <hip/hip_runtime.h>
#include <hip/hip_bf16.h>

typedef __attribute__((ext_vector_type(16))) _Float16 v16h;
typedef __attribute__((ext_vector_type(8)))  float    v8f;
typedef __attribute__((ext_vector_type(4)))  float    v4f;

#define FIN 64
#define HID 256
#define NLY 2
#define TT  32
#define MN  128
#define KC  4
#define SS  64
#define DD  4096   // T*M
#define G4  1024   // 4*HID

// ---------------- workspace layout (bytes) ----------------
constexpr size_t OFF_XH   = 0;                          // x f16            [8192][64]
constexpr size_t OFF_AH   = OFF_XH   + 1048576;         // A_norm f16       [128][128]
constexpr size_t OFF_WGT  = OFF_AH   + 32768;           // W_gcn f16        [256][64]   (B as [N][K])
constexpr size_t OFF_WIH  = OFF_WGT  + 32768;           // Wih f16          [2][1024][256]
constexpr size_t OFF_WHH  = OFF_WIH  + 1048576;         // Whh f16          [2][1024][256]
constexpr size_t OFF_BSUM = OFF_WHH  + 1048576;         // bih+bhh f32      [2][1024]
constexpr size_t OFF_XWT  = OFF_BSUM + 8192;            // XwT f16          [16384][128] (B as [N][K])
constexpr size_t OFF_HSEQ = OFF_XWT  + 4194304;         // time-major f16   [64][128][256]
constexpr size_t OFF_SEQ1 = OFF_HSEQ + 4194304;         // layer0 out f16   [64][128][256]
constexpr size_t OFF_G    = OFF_SEQ1 + 4194304;         // gates f32        [128][1024]
constexpr size_t OFF_CST  = OFF_G    + 524288;          // cell f32         [128][256]
constexpr size_t OFF_HFF  = OFF_CST  + 131072;          // h_final f32      [128][256]
constexpr size_t OFF_HFH  = OFF_HFF  + 131072;          // h_final f16
constexpr size_t OFF_TEMW = OFF_HFH  + 65536;           // tem_W f16        [4][32][256]
constexpr size_t OFF_SPAW = OFF_TEMW + 65536;           // spa_W f16        [4][128][256]
constexpr size_t OFF_TB   = OFF_SPAW + 262144;          // t f32            [4][128][32]
constexpr size_t OFF_SB   = OFF_TB   + 65536;           // s f32            [4][128][128]
constexpr size_t OFF_DV   = OFF_SB   + 262144;          // d f32            [4][128]
constexpr size_t OFF_TEMG = OFF_DV   + 2048;            // tem f32          [4][32][32]
constexpr size_t OFF_SPAG = OFF_TEMG + 16384;           // spa f32          [4][128][128]
constexpr size_t OFF_VKH  = OFF_SPAG + 262144;          // v f32            [4][256]
constexpr size_t OFF_WSUM = OFF_VKH  + 4096;            // wsum f32         [4]

// ---------------- WMMA fragment helpers (ISA 7.12.2 layouts) ----------------
// A operand: [M][K] row-major. lane holds row (lane&15); per-lane data contiguous in K.
__device__ __forceinline__ v16h load_frag_a(const _Float16* A, int lda, int row0, int k0, int lane) {
    int r  = row0 + (lane & 15);
    int kb = k0 + ((lane >> 4) << 3);        // 0 or 8
    const _Float16* p = A + (long)r * lda + kb;
    v16h f;
#pragma unroll
    for (int e = 0; e < 8; ++e) { f[e] = p[e]; f[8 + e] = p[16 + e]; }
    return f;
}

// B operand stored [N][K] ("column-major" for the GEMM): per-lane data contiguous in K,
// so each fragment is two 16-byte loads instead of a 16-element gather.
__device__ __forceinline__ v16h load_frag_b(const _Float16* B, int ldb, int k0, int col0, int lane) {
    int nr = col0 + (lane & 15);
    int kb = k0 + ((lane >> 4) << 3);
    const _Float16* p = B + (long)nr * ldb + kb;
    v16h f;
#pragma unroll
    for (int e = 0; e < 8; ++e) { f[e] = p[e]; f[8 + e] = p[16 + e]; }
    return f;
}

__device__ __forceinline__ v8f wmma_f16(v16h a, v16h b, v8f c) {
    return __builtin_amdgcn_wmma_f32_16x16x32_f16(false, a, false, b, (short)0, c, false, false);
}

__device__ __forceinline__ float sigf(float x) { return 1.0f / (1.0f + __expf(-x)); }

// ---------------- prep kernels ----------------
__global__ void k_cvt_f16(const float* in, _Float16* out, int n) {
    int i = blockIdx.x * blockDim.x + threadIdx.x;
    if (i < n) out[i] = (_Float16)in[i];
}

__global__ void k_bias_sum(const float* a, const float* b, float* o, int n) {
    int i = blockIdx.x * blockDim.x + threadIdx.x;
    if (i < n) o[i] = a[i] + b[i];
}

// ---------------- generic batched WMMA GEMM: C = act(A@B^T + bias), B stored [N][K] ----------------
// cmode 0: C[row][col] (ld = Nc)
// cmode 1: GCN1 special: row = m*SS+t, store at [(t*HID+col)][m]   (-> XwT, B-layout for GEMM2)
// cmode 2: GCN2 special: col = t*HID+h, store at [t][row][h]       (-> time-major LSTM input)
__global__ void k_gemm(const _Float16* A, long strideA,
                       const _Float16* B, long strideB,
                       const float* bias, long strideBias,
                       float* Cf, _Float16* Ch, long strideC,
                       int Mr, int Nc, int Kd, int act, int cmode) {
    int lane = threadIdx.x & 31;
    int wave = (blockIdx.x * blockDim.x + threadIdx.x) >> 5;
    int tilesN = Nc >> 4, tilesM = Mr >> 4;
    if (wave >= tilesM * tilesN) return;
    int tm = wave / tilesN, tn = wave % tilesN;
    const _Float16* Ab = A + (long)blockIdx.y * strideA;
    const _Float16* Bb = B + (long)blockIdx.y * strideB;
    v8f acc = {};
    for (int k = 0; k < Kd; k += 32) {
        if (k + 32 < Kd) {   // gfx1250 global_prefetch of next K-chunk
            __builtin_prefetch(Ab + (long)(tm * 16 + (lane & 15)) * Kd + k + 32, 0, 1);
            __builtin_prefetch(Bb + (long)(tn * 16 + (lane & 15)) * Kd + k + 32, 0, 1);
        }
        v16h a = load_frag_a(Ab, Kd, tm * 16, k, lane);
        v16h b = load_frag_b(Bb, Kd, k, tn * 16, lane);
        acc = wmma_f16(a, b, acc);
    }
    int n = tn * 16 + (lane & 15);
    float bv = bias ? bias[(long)blockIdx.y * strideBias + n] : 0.0f;
    long cbase = (long)blockIdx.y * strideC;
#pragma unroll
    for (int v = 0; v < 8; ++v) {
        int m = tm * 16 + ((lane >> 4) << 3) + v;
        float val = acc[v] + bv;
        if (act) val = fmaxf(val, 0.0f);
        long ci;
        if (cmode == 0)      ci = cbase + (long)m * Nc + n;
        else if (cmode == 1) ci = ((long)(m & (SS - 1)) * HID + n) * MN + (m >> 6);
        else                 ci = (long)(n >> 8) * (MN * HID) + (long)m * HID + (n & 255);
        if (Cf) Cf[ci] = val;
        if (Ch) Ch[ci] = (_Float16)val;
    }
}

// ---------------- LSTM: single workgroup (32 wave32), H state resident in LDS ----------------
__global__ void __launch_bounds__(1024)
k_lstm(const _Float16* seq0, _Float16* seq1,
       const _Float16* Wih, const _Float16* Whh, const float* bsum,
       float* G, float* Cst, float* hfin_f, _Float16* hfin_h) {
    __shared__ _Float16 Hsh[MN * HID];   // 64 KB of the WGP's 320 KB LDS
    int tid = threadIdx.x, lane = tid & 31, wave = tid >> 5;   // 32 waves
    for (int l = 0; l < NLY; ++l) {
        for (int i = tid; i < MN * HID; i += 1024) { Cst[i] = 0.0f; Hsh[i] = (_Float16)0.0f; }
        __threadfence(); __syncthreads();
        const _Float16* Wx = Wih + (long)l * G4 * HID;   // [1024][256]  B as [N][K]
        const _Float16* Wh = Whh + (long)l * G4 * HID;
        const float*    bl = bsum + l * G4;
        const _Float16* in_seq = (l == 0) ? seq0 : seq1;
        for (int t = 0; t < SS; ++t) {
            const _Float16* Xt = in_seq + (long)t * MN * HID;
            // G = Xt@Wx^T + H@Wh^T + b : 8x64 = 512 tiles, 16 per wave
            for (int ti = 0; ti < 16; ++ti) {
                int tile = wave * 16 + ti;
                int tm = tile >> 6, tn = tile & 63;
                v8f acc = {};
                for (int k = 0; k < HID; k += 32) {
                    v16h a = load_frag_a(Xt, HID, tm * 16, k, lane);
                    v16h b = load_frag_b(Wx, HID, k, tn * 16, lane);
                    acc = wmma_f16(a, b, acc);
                    a = load_frag_a(Hsh, HID, tm * 16, k, lane);   // ds_load from LDS
                    b = load_frag_b(Wh, HID, k, tn * 16, lane);
                    acc = wmma_f16(a, b, acc);
                }
                int n = tn * 16 + (lane & 15);
                float bv = bl[n];
#pragma unroll
                for (int v = 0; v < 8; ++v) {
                    int m = tm * 16 + ((lane >> 4) << 3) + v;
                    G[(long)m * G4 + n] = acc[v] + bv;
                }
            }
            __threadfence(); __syncthreads();
            // cell update (PyTorch gate order i,f,g,o)
            for (int i = tid; i < MN * HID; i += 1024) {
                int m = i >> 8, h = i & 255;
                const float* gr = G + (long)m * G4;
                float gi = gr[h], gf = gr[256 + h], gg = gr[512 + h], go = gr[768 + h];
                float c = sigf(gf) * Cst[i] + sigf(gi) * tanhf(gg);
                float hv = sigf(go) * tanhf(c);
                Cst[i] = c;
                Hsh[i] = (_Float16)hv;
                if (l == 0) seq1[(long)t * MN * HID + i] = (_Float16)hv;
                if (l == 1 && t == SS - 1) { hfin_f[i] = hv; hfin_h[i] = (_Float16)hv; }
            }
            __threadfence(); __syncthreads();
        }
    }
}

// ---------------- heads ----------------
__global__ void k_vkh(const float* hfin, const float* w2, float* v, float* wsum) {
    int k = blockIdx.x, h = threadIdx.x;
    float acc = 0.0f;
    for (int m = 0; m < MN; ++m) acc += w2[k * MN + m] * hfin[m * HID + h];
    v[k * HID + h] = acc;
    if (h == 0) { float s = 0.0f; for (int m = 0; m < MN; ++m) s += w2[k * MN + m]; wsum[k] = s; }
}

__global__ void k_mu(const float* v, const float* wsum, const float* W1, const float* b1,
                     const float* b2, float* mu) {
    int i = blockIdx.x * blockDim.x + threadIdx.x;   // k*D + d
    if (i >= KC * DD) return;
    int k = i >> 12;
    const float* w  = W1 + (long)i * HID;
    const float* vk = v + k * HID;
    float acc = 0.0f;
    for (int h = 0; h < HID; ++h) acc += vk[h] * w[h];
    mu[i] = acc + b1[i] * wsum[k] + b2[k];
}

__global__ void k_pi(const float* hfin, const float* p1W, const float* p1b,
                     const float* p2W, const float* p2b, float* out) {
    __shared__ float sm[MN];
    __shared__ float lg[KC];
    int m = threadIdx.x;
    for (int k = 0; k < KC; ++k) {
        float p = p1b[k];
        for (int h = 0; h < HID; ++h) p += hfin[m * HID + h] * p1W[k * HID + h];
        sm[m] = p2W[m] * p;
        __syncthreads();
        for (int s = 64; s > 0; s >>= 1) { if (m < s) sm[m] += sm[m + s]; __syncthreads(); }
        if (m == 0) lg[k] = sm[0] + p2b[0];
        __syncthreads();
    }
    if (m == 0) {
        float mx = fmaxf(fmaxf(lg[0], lg[1]), fmaxf(lg[2], lg[3]));
        float e0 = __expf(lg[0] - mx), e1 = __expf(lg[1] - mx);
        float e2 = __expf(lg[2] - mx), e3 = __expf(lg[3] - mx);
        float s = e0 + e1 + e2 + e3;
        out[0] = e0 / s; out[1] = e1 / s; out[2] = e2 / s; out[3] = e3 / s;
    }
}

__global__ void k_d(const float* hfin, const float* dW, const float* db, float* dvec) {
    int i = blockIdx.x * blockDim.x + threadIdx.x;
    if (i >= KC * MN) return;
    int k = i >> 7, m = i & 127;
    float acc = db[k];
    for (int h = 0; h < HID; ++h) acc += hfin[m * HID + h] * dW[k * HID + h];
    float sp = (acc > 20.0f) ? acc : log1pf(__expf(acc));
    dvec[i] = sp + 1e-4f;
}

// out[k][a][b] = sum_m in[k][m][a]*in[k][m][b]
__global__ void k_gram(const float* in, float* out, int Ncol) {
    int i = blockIdx.x * blockDim.x + threadIdx.x;
    if (i >= Ncol * Ncol) return;
    int a = i / Ncol, b = i % Ncol;
    const float* base = in + (long)blockIdx.y * MN * Ncol;
    float acc = 0.0f;
    for (int m = 0; m < MN; ++m) acc += base[m * Ncol + a] * base[m * Ncol + b];
    out[(long)blockIdx.y * Ncol * Ncol + i] = acc;
}

// cov[k][d1][d2] = tem[k][tt][uu]*spa[k][a][b] (+ diag) -- 268 MB streamed with NT stores
__global__ void k_cov(const float* tem, const float* spa, const float* dvec, float* cov) {
    long total = (long)KC * DD * DD / 4;
    long stride = (long)gridDim.x * blockDim.x;
    for (long idx = (long)blockIdx.x * blockDim.x + threadIdx.x; idx < total; idx += stride) {
        long e = idx * 4;
        int k  = (int)(e / ((long)DD * DD));
        long r = e % ((long)DD * DD);
        int d1 = (int)(r / DD), d2 = (int)(r % DD);   // d2 multiple of 4
        int tt = d1 >> 7, a = d1 & 127;
        int uu = d2 >> 7, b = d2 & 127;
        float tv = tem[(k * TT + tt) * TT + uu];
        v4f sp = *(const v4f*)&spa[((long)k * MN + a) * MN + b];
        v4f o = tv * sp;
        if (tt == uu && a >= b && a < b + 4) {
            float add = dvec[k * MN + a] + 1e-6f;
            o[a - b] += add;
        }
        __builtin_nontemporal_store(o, (v4f*)&cov[e]);
    }
}

// ---------------- launch ----------------
extern "C" void kernel_launch(void* const* d_in, const int* in_sizes, int n_in,
                              void* d_out, int out_size, void* d_ws, size_t ws_size,
                              hipStream_t stream) {
    const float* x       = (const float*)d_in[0];
    const float* A_norm  = (const float*)d_in[1];
    const float* W_gcn   = (const float*)d_in[2];
    const float* b_gcn   = (const float*)d_in[3];
    const float* Wih     = (const float*)d_in[4];
    const float* Whh     = (const float*)d_in[5];
    const float* bih     = (const float*)d_in[6];
    const float* bhh     = (const float*)d_in[7];
    const float* pi1W    = (const float*)d_in[8];
    const float* pi1b    = (const float*)d_in[9];
    const float* pi2W    = (const float*)d_in[10];
    const float* pi2b    = (const float*)d_in[11];
    const float* temW    = (const float*)d_in[12];
    const float* temb    = (const float*)d_in[13];
    const float* spaW    = (const float*)d_in[14];
    const float* spab    = (const float*)d_in[15];
    const float* diagW   = (const float*)d_in[16];
    const float* diagb   = (const float*)d_in[17];
    const float* m1W     = (const float*)d_in[18];
    const float* m1b     = (const float*)d_in[19];
    const float* m2W     = (const float*)d_in[20];
    const float* m2b     = (const float*)d_in[21];

    char* ws = (char*)d_ws;
    _Float16* x_h    = (_Float16*)(ws + OFF_XH);
    _Float16* A_h    = (_Float16*)(ws + OFF_AH);
    _Float16* Wgcn_h = (_Float16*)(ws + OFF_WGT);
    _Float16* Wih_h  = (_Float16*)(ws + OFF_WIH);
    _Float16* Whh_h  = (_Float16*)(ws + OFF_WHH);
    float*    bsum   = (float*)(ws + OFF_BSUM);
    _Float16* XwT_h  = (_Float16*)(ws + OFF_XWT);
    _Float16* Hseq_h = (_Float16*)(ws + OFF_HSEQ);
    _Float16* seq1_h = (_Float16*)(ws + OFF_SEQ1);
    float*    Gbuf   = (float*)(ws + OFF_G);
    float*    Cst    = (float*)(ws + OFF_CST);
    float*    hfin_f = (float*)(ws + OFF_HFF);
    _Float16* hfin_h = (_Float16*)(ws + OFF_HFH);
    _Float16* temW_h = (_Float16*)(ws + OFF_TEMW);
    _Float16* spaW_h = (_Float16*)(ws + OFF_SPAW);
    float*    t_buf  = (float*)(ws + OFF_TB);
    float*    s_buf  = (float*)(ws + OFF_SB);
    float*    dvec   = (float*)(ws + OFF_DV);
    float*    temG   = (float*)(ws + OFF_TEMG);
    float*    spaG   = (float*)(ws + OFF_SPAG);
    float*    vkh    = (float*)(ws + OFF_VKH);
    float*    wsum   = (float*)(ws + OFF_WSUM);

    float* out = (float*)d_out;
    float* pi_out  = out;
    float* mu_out  = out + 4;
    float* cov_out = out + 4 + KC * DD;

    // --- prep: f32 -> f16 (weights are naturally [N][K] = B layout; no transposes needed) ---
    k_cvt_f16<<<(MN*SS*FIN + 255) / 256, 256, 0, stream>>>(x, x_h, MN * SS * FIN);
    k_cvt_f16<<<(MN*MN + 255) / 256, 256, 0, stream>>>(A_norm, A_h, MN * MN);
    k_cvt_f16<<<(HID*FIN + 255) / 256, 256, 0, stream>>>(W_gcn, Wgcn_h, HID * FIN);
    k_cvt_f16<<<(NLY*G4*HID + 255) / 256, 256, 0, stream>>>(Wih, Wih_h, NLY * G4 * HID);
    k_cvt_f16<<<(NLY*G4*HID + 255) / 256, 256, 0, stream>>>(Whh, Whh_h, NLY * G4 * HID);
    k_cvt_f16<<<(KC*TT*HID + 255) / 256, 256, 0, stream>>>(temW, temW_h, KC * TT * HID);
    k_cvt_f16<<<(KC*MN*HID + 255) / 256, 256, 0, stream>>>(spaW, spaW_h, KC * MN * HID);
    k_bias_sum<<<(NLY*G4 + 255) / 256, 256, 0, stream>>>(bih, bhh, bsum, NLY * G4);

    // --- GCN1: Xw = x@W_gcn^T + b -> stored transposed [th][m] (B layout for GCN2) ---
    k_gemm<<<dim3(1024, 1), 256, 0, stream>>>(x_h, 0, Wgcn_h, 0, b_gcn, 0,
                                              nullptr, XwT_h, 0, MN * SS, HID, FIN, 0, 1);
    // --- GCN2: H = relu(A_norm @ Xw) -> stored time-major [t][m][h] ---
    k_gemm<<<dim3(1024, 1), 256, 0, stream>>>(A_h, 0, XwT_h, 0, nullptr, 0,
                                              nullptr, Hseq_h, 0, MN, SS * HID, MN, 1, 2);

    // --- LSTM (persistent single-WGP workgroup, WMMA gates, H in LDS) ---
    k_lstm<<<1, 1024, 0, stream>>>(Hseq_h, seq1_h, Wih_h, Whh_h, bsum,
                                   Gbuf, Cst, hfin_f, hfin_h);

    // --- mixture heads ---
    k_pi<<<1, MN, 0, stream>>>(hfin_f, pi1W, pi1b, pi2W, pi2b, pi_out);
    k_vkh<<<KC, HID, 0, stream>>>(hfin_f, m2W, vkh, wsum);
    k_mu<<<(KC*DD + 255) / 256, 256, 0, stream>>>(vkh, wsum, m1W, m1b, m2b, mu_out);
    // t[k] = h@temW[k]^T + b   [128][32] per k
    k_gemm<<<dim3(2, KC), 256, 0, stream>>>(hfin_h, 0, temW_h, (long)TT * HID, temb, TT,
                                            t_buf, nullptr, (long)MN * TT, MN, TT, HID, 0, 0);
    // s[k] = h@spaW[k]^T + b   [128][128] per k
    k_gemm<<<dim3(8, KC), 256, 0, stream>>>(hfin_h, 0, spaW_h, (long)MN * HID, spab, MN,
                                            s_buf, nullptr, (long)MN * MN, MN, MN, HID, 0, 0);
    k_d<<<(KC*MN + 255) / 256, 256, 0, stream>>>(hfin_f, diagW, diagb, dvec);

    // --- grams ---
    k_gram<<<dim3((TT*TT + 255) / 256, KC), 256, 0, stream>>>(t_buf, temG, TT);
    k_gram<<<dim3((MN*MN + 255) / 256, KC), 256, 0, stream>>>(s_buf, spaG, MN);

    // --- cov = kron(tem, spa) + diag : 268 MB streaming NT writes ---
    k_cov<<<16384, 256, 0, stream>>>(temG, spaG, dvec, cov_out);
}
